// SelfAttention_9234179687076
// MI455X (gfx1250) — matrix operand order repaired and checked by reading, not compiled, not run on hardware
//
#include <hip/hip_runtime.h>
#include <hip/hip_bf16.h>

typedef __attribute__((ext_vector_type(16))) _Float16 v16h;
typedef __attribute__((ext_vector_type(8)))  _Float16 v8h_t;
typedef __attribute__((ext_vector_type(8)))  float    v8f;

#define SEQ   4096
#define HID   128
#define NHD   8
#define DPROJ (HID*NHD)   // 1024

// ---- fragment loaders -------------------------------------------------------
// A (16x32 f16), row-major source. CDNA5 layout:
//   lanes 0-15 : row=lane,    K {0..7, 16..23}
//   lanes 16-31: row=lane-16, K {8..15, 24..31}
static __device__ __forceinline__ v16h loadA(const _Float16* rowBase, int ld) {
  const int lane = threadIdx.x & 31;
  const int hl   = lane >> 4;
  const _Float16* p = rowBase + (size_t)(lane & 15) * ld;
  v8h_t lo = *(const v8h_t*)(p + (hl ? 8  : 0));
  v8h_t hi = *(const v8h_t*)(p + (hl ? 24 : 16));
  v16h a;
#pragma unroll
  for (int i = 0; i < 8; ++i) { a[i] = lo[i]; a[i + 8] = hi[i]; }
  return a;
}

// B (32x16 f16) from "B-transposed" storage BT[col][k]:
//   lanes 0-15 : col=lane,    K {0..7, 16..23}
//   lanes 16-31: col=lane-16, K {8..15, 24..31}
static __device__ __forceinline__ v16h loadB(const _Float16* btBase, int ld) {
  const int lane = threadIdx.x & 31;
  const int hl   = lane >> 4;
  const _Float16* p = btBase + (size_t)(lane & 15) * ld;
  v8h_t lo = *(const v8h_t*)(p + 8 * hl);
  v8h_t hi = *(const v8h_t*)(p + 16 + 8 * hl);
  v16h b;
#pragma unroll
  for (int i = 0; i < 8; ++i) { b[i] = lo[i]; b[i + 8] = hi[i]; }
  return b;
}

// reductions across a 16-lane half (C-layout rows live within a half)
static __device__ __forceinline__ float redMax16(float v) {
#pragma unroll
  for (int m = 1; m < 16; m <<= 1) v = fmaxf(v, __shfl_xor(v, m, 32));
  return v;
}
static __device__ __forceinline__ float redSum16(float v) {
#pragma unroll
  for (int m = 1; m < 16; m <<= 1) v += __shfl_xor(v, m, 32);
  return v;
}

// ---- prep: f32->f16 convert + transposes -----------------------------------
__global__ void prep_kernel(const float* __restrict__ x, const float* __restrict__ W1,
                            const float* __restrict__ W2, const float* __restrict__ Wc,
                            _Float16* __restrict__ xh, _Float16* __restrict__ W1t,
                            _Float16* __restrict__ W2t, _Float16* __restrict__ Wct) {
  const int NX = SEQ * HID;      // 524288
  const int NW = DPROJ * HID;    // 131072
  const int total = NX + 3 * NW;
  for (int i = blockIdx.x * blockDim.x + threadIdx.x; i < total;
       i += gridDim.x * blockDim.x) {
    if (i < NX) {
      xh[i] = (_Float16)x[i];
    } else if (i < NX + NW) {                       // W1t[j][k] = W1[k][j]
      int t = i - NX; int j = t / HID, k = t % HID;
      W1t[t] = (_Float16)W1[(size_t)k * DPROJ + j];
    } else if (i < NX + 2 * NW) {                   // W2t[j][k] = W2[k][j]
      int t = i - NX - NW; int j = t / HID, k = t % HID;
      W2t[t] = (_Float16)W2[(size_t)k * DPROJ + j];
    } else {                                        // Wct[n][k] = Wc[k][n]
      int t = i - NX - 2 * NW; int n = t / DPROJ, k = t % DPROJ;
      Wct[t] = (_Float16)Wc[(size_t)k * HID + n];
    }
  }
}

// ---- q1 = x@W1+b1, q2 = x@W2+b2; emit per-head row-major + V-transposed ----
__global__ __launch_bounds__(128)
void proj_kernel(const _Float16* __restrict__ xh, const _Float16* __restrict__ W1t,
                 const _Float16* __restrict__ W2t, const float* __restrict__ b1,
                 const float* __restrict__ b2, _Float16* __restrict__ q1h,
                 _Float16* __restrict__ q2h, _Float16* __restrict__ q2t) {
  const int w    = threadIdx.x >> 5;
  const int lane = threadIdx.x & 31;
  const int hl   = lane >> 4;
  const int colL = lane & 15;
  const int h    = blockIdx.y;
  const int m0   = blockIdx.x * 16;

  v16h ax[4];
#pragma unroll
  for (int kc = 0; kc < 4; ++kc) ax[kc] = loadA(xh + (size_t)m0 * HID + kc * 32, HID);

#pragma unroll
  for (int nt = 0; nt < 2; ++nt) {
    const int nG0 = h * HID + w * 32 + nt * 16;   // global output column base
    v8f a1 = {}, a2 = {};
#pragma unroll
    for (int kc = 0; kc < 4; ++kc) {
      v16h bw1 = loadB(W1t + (size_t)nG0 * HID + kc * 32, HID);
      v16h bw2 = loadB(W2t + (size_t)nG0 * HID + kc * 32, HID);
      a1 = __builtin_amdgcn_wmma_f32_16x16x32_f16(false, ax[kc], false, bw1, (short)0, a1, false, false);
      a2 = __builtin_amdgcn_wmma_f32_16x16x32_f16(false, ax[kc], false, bw2, (short)0, a2, false, false);
    }
    const int nG  = nG0 + colL;
    const int cIn = nG - h * HID;                 // column within head
    const float bb1 = b1[nG], bb2 = b2[nG];
#pragma unroll
    for (int r = 0; r < 8; ++r) {
      const int s = m0 + r + 8 * hl;
      _Float16 v1 = (_Float16)(a1[r] + bb1);
      _Float16 v2 = (_Float16)(a2[r] + bb2);
      q1h[((size_t)h * SEQ + s) * HID + cIn] = v1;
      q2h[((size_t)h * SEQ + s) * HID + cIn] = v2;
      q2t[((size_t)h * HID + cIn) * SEQ + s] = v2;   // V^T for ctx B-frags
    }
  }
}

// ---- flash attention: ctx = softmax(q1 q2^T) q2, per head ------------------
__global__ __launch_bounds__(128)
void attn_kernel(const _Float16* __restrict__ q1h, const _Float16* __restrict__ q2h,
                 const _Float16* __restrict__ q2t, _Float16* __restrict__ ctxh) {
  __shared__ _Float16 ldsP[4][16][40];            // per-wave 16x32 P tile, padded
  const int w    = threadIdx.x >> 5;
  const int lane = threadIdx.x & 31;
  const int hl   = lane >> 4;
  const int colL = lane & 15;
  const int h    = blockIdx.y;
  const int m0   = blockIdx.x * 64 + w * 16;      // 16 query rows per wave

  const _Float16* Q1 = q1h + ((size_t)h * SEQ + m0) * HID;
  const _Float16* Kp = q2h + (size_t)h * SEQ * HID;
  const _Float16* Vt = q2t + (size_t)h * HID * SEQ;

  v16h aq[4];
#pragma unroll
  for (int kc = 0; kc < 4; ++kc) aq[kc] = loadA(Q1 + kc * 32, HID);

  v8f acc[8];
#pragma unroll
  for (int nt = 0; nt < 8; ++nt) acc[nt] = (v8f){};
  float mrow[8], lrow[8];
#pragma unroll
  for (int r = 0; r < 8; ++r) { mrow[r] = -1e30f; lrow[r] = 0.f; }

  for (int j0 = 0; j0 < SEQ; j0 += 32) {
    // --- issue ALL chunk loads up front into distinct registers so waits
    //     stagger instead of serializing each wmma on loadcnt==0 ---
    v16h bk[8];                                   // K frags: 2 key tiles x 4 kc
#pragma unroll
    for (int kc = 0; kc < 4; ++kc) {
      bk[kc]     = loadB(Kp + (size_t)j0 * HID + kc * 32, HID);
      bk[4 + kc] = loadB(Kp + (size_t)(j0 + 16) * HID + kc * 32, HID);
    }
    v16h bv[8];                                   // V frags (independent of scores;
#pragma unroll                                    //  overlap softmax VALU work)
    for (int nt = 0; nt < 8; ++nt)
      bv[nt] = loadB(Vt + (size_t)(nt * 16) * SEQ + j0, SEQ);

    // prefetch next chunk while this one computes (global_prefetch_b8)
    {
      const _Float16* pk = Kp + (size_t)(j0 + 32 + lane) * HID;
      const _Float16* pv = Vt + (size_t)(lane * 4) * SEQ + (j0 + 32);
      __builtin_prefetch(pk, 0, 1);
      __builtin_prefetch(pk + 64, 0, 1);          // second 128B of the 256B row
      __builtin_prefetch(pv, 0, 1);
    }

    // scores for 32 keys: two 16x16 C tiles
    v8f s0 = {}, s1 = {};
#pragma unroll
    for (int kc = 0; kc < 4; ++kc) {
      s0 = __builtin_amdgcn_wmma_f32_16x16x32_f16(false, aq[kc], false, bk[kc],     (short)0, s0, false, false);
      s1 = __builtin_amdgcn_wmma_f32_16x16x32_f16(false, aq[kc], false, bk[4 + kc], (short)0, s1, false, false);
    }
    // online softmax (per-row stats replicated across each 16-lane half)
    float p0[8], p1[8];
#pragma unroll
    for (int r = 0; r < 8; ++r) {
      float t  = redMax16(fmaxf(s0[r], s1[r]));
      float nm = fmaxf(mrow[r], t);
      float al = __expf(mrow[r] - nm);
      mrow[r]  = nm;
      p0[r] = __expf(s0[r] - nm);
      p1[r] = __expf(s1[r] - nm);
      lrow[r] = lrow[r] * al + redSum16(p0[r] + p1[r]);
#pragma unroll
      for (int nt = 0; nt < 8; ++nt) acc[nt][r] *= al;
    }
    // C-layout -> A-layout via per-wave LDS tile
#pragma unroll
    for (int r = 0; r < 8; ++r) {
      const int row = r + 8 * hl;
      ldsP[w][row][colL]      = (_Float16)p0[r];
      ldsP[w][row][colL + 16] = (_Float16)p1[r];
    }
    asm volatile("s_wait_dscnt 0" ::: "memory");
    v16h pa = loadA(&ldsP[w][0][0], 40);
    // ctx += P (16x32) x V (32x16 per n-tile); bv already resident
#pragma unroll
    for (int nt = 0; nt < 8; ++nt)
      acc[nt] = __builtin_amdgcn_wmma_f32_16x16x32_f16(false, pa, false, bv[nt], (short)0, acc[nt], false, false);
  }
  // normalize and store ctx as f16: ctxh[s][h*128 + n]
#pragma unroll
  for (int r = 0; r < 8; ++r) {
    const float inv = 1.f / lrow[r];
    const int s = m0 + r + 8 * hl;
    _Float16* dst = ctxh + (size_t)s * DPROJ + h * HID + colL;
#pragma unroll
    for (int nt = 0; nt < 8; ++nt) dst[nt * 16] = (_Float16)(acc[nt][r] * inv);
  }
}

// ---- out = ctx @ Wc + bc + x ----------------------------------------------
__global__ __launch_bounds__(128)
void outproj_kernel(const _Float16* __restrict__ ctxh, const _Float16* __restrict__ Wct,
                    const float* __restrict__ bc, const float* __restrict__ x,
                    float* __restrict__ out) {
  const int w    = threadIdx.x >> 5;
  const int lane = threadIdx.x & 31;
  const int hl   = lane >> 4;
  const int colL = lane & 15;
  const int nt   = blockIdx.y * 4 + w;
  const int n0   = nt * 16;
  const int m0   = blockIdx.x * 16;

  v8f acc = {};
#pragma unroll 4
  for (int kc = 0; kc < 32; ++kc) {
    v16h a = loadA(ctxh + (size_t)m0 * DPROJ + kc * 32, DPROJ);
    v16h b = loadB(Wct + (size_t)n0 * DPROJ + kc * 32, DPROJ);
    acc = __builtin_amdgcn_wmma_f32_16x16x32_f16(false, a, false, b, (short)0, acc, false, false);
  }
  const int n = n0 + colL;
  const float bcv = bc[n];
#pragma unroll
  for (int r = 0; r < 8; ++r) {
    const int s = m0 + r + 8 * hl;
    out[(size_t)s * HID + n] = acc[r] + bcv + x[(size_t)s * HID + n];
  }
}

// ---- launcher ---------------------------------------------------------------
extern "C" void kernel_launch(void* const* d_in, const int* in_sizes, int n_in,
                              void* d_out, int out_size, void* d_ws, size_t ws_size,
                              hipStream_t stream) {
  (void)in_sizes; (void)n_in; (void)out_size; (void)ws_size;
  const float* x  = (const float*)d_in[0];
  const float* W1 = (const float*)d_in[1];
  const float* b1 = (const float*)d_in[2];
  const float* W2 = (const float*)d_in[3];
  const float* b2 = (const float*)d_in[4];
  const float* Wc = (const float*)d_in[5];
  const float* bc = (const float*)d_in[6];
  float* out = (float*)d_out;

  char* p = (char*)d_ws;
  auto carve = [&](size_t elems) -> _Float16* {
    _Float16* r = (_Float16*)p;
    p += ((elems * 2 + 255) / 256) * 256;
    return r;
  };
  _Float16* xh   = carve((size_t)SEQ * HID);
  _Float16* W1t  = carve((size_t)DPROJ * HID);
  _Float16* W2t  = carve((size_t)DPROJ * HID);
  _Float16* Wct  = carve((size_t)DPROJ * HID);
  _Float16* q1h  = carve((size_t)NHD * SEQ * HID);
  _Float16* q2h  = carve((size_t)NHD * SEQ * HID);
  _Float16* q2t  = carve((size_t)NHD * SEQ * HID);
  _Float16* ctxh = carve((size_t)SEQ * DPROJ);

  prep_kernel<<<512, 256, 0, stream>>>(x, W1, W2, Wc, xh, W1t, W2t, Wct);
  dim3 gp(SEQ / 16, NHD);
  proj_kernel<<<gp, 128, 0, stream>>>(xh, W1t, W2t, b1, b2, q1h, q2h, q2t);
  dim3 ga(SEQ / 64, NHD);
  attn_kernel<<<ga, 128, 0, stream>>>(q1h, q2h, q2t, ctxh);
  dim3 go(SEQ / 16, 2);
  outproj_kernel<<<go, 128, 0, stream>>>(ctxh, Wct, bc, x, out);
}
